// VN_PCTransformer_47313359733181
// MI455X (gfx1250) — matrix operand order repaired and checked by reading, not compiled, not run on hardware
//
#include <hip/hip_runtime.h>
#include <hip/hip_bf16.h>

typedef _Float16 h8   __attribute__((ext_vector_type(8)));
typedef _Float16 v16h __attribute__((ext_vector_type(16)));
typedef float    v8f  __attribute__((ext_vector_type(8)));
typedef unsigned int ui4 __attribute__((ext_vector_type(4)));
typedef unsigned int ui8 __attribute__((ext_vector_type(8)));

#define B_    4
#define N_    2048
#define C_    128
#define D_    384
#define H_    6
#define KNN_  8
#define HD_   192
#define EPS_   1e-6f
#define LNEPS_ 1e-5f
#define NS_    0.2f

static __device__ __forceinline__ v8f wmma_f16(v16h a, v16h b, v8f c) {
  return __builtin_amdgcn_wmma_f32_16x16x32_f16(false, a, false, b, (short)0, c, false, false);
}

static __device__ __forceinline__ v16h cat16(h8 a, h8 b) {
  return __builtin_shufflevector(a, b, 0,1,2,3,4,5,6,7,8,9,10,11,12,13,14,15);
}

// A-operand fragment (16 rows x 32 K), rows at base + (lane&15)*ld, K chunk at c0.
static __device__ __forceinline__ v16h afrag(const _Float16* base, int ld, int c0, int lane) {
  const _Float16* r = base + (size_t)(lane & 15) * ld + c0 + ((lane >> 4) << 3);
  h8 x0 = *(const h8*)(r);
  h8 x1 = *(const h8*)(r + 16);
  return cat16(x0, x1);
}

// B-operand fragment (32 K x 16 cols): col = lane&15, elem e -> K = (l>>4)*8+e (e<8), 16+(l>>4)*8+(e-8) (e>=8).
static __device__ __forceinline__ v16h bfrag(const _Float16* base, int ld, int c0, int lane) {
  const _Float16* r = base + (size_t)(lane & 15) * ld + c0 + ((lane >> 4) << 3);
  h8 x0 = *(const h8*)(r);
  h8 x1 = *(const h8*)(r + 16);
  return cat16(x0, x1);
}

// ---------------------------------------------------------------- weight cvt
__global__ void k_cvt(const float* __restrict__ s, _Float16* __restrict__ d, int n) {
  for (int i = blockIdx.x * blockDim.x + threadIdx.x; i < n; i += gridDim.x * blockDim.x)
    d[i] = (_Float16)s[i];
}

// ---------------------------------------------------------------- LayerNorm 1
__global__ void k_ln1(const float* __restrict__ x, const float* __restrict__ g,
                      const float* __restrict__ bb, _Float16* __restrict__ nx) {
  int bn = blockIdx.x, c = threadIdx.x;
  const float* p = x + (size_t)bn * D_ + c * 3;
  float x0 = p[0], x1 = p[1], x2 = p[2];
  float nc = sqrtf(x0*x0 + x1*x1 + x2*x2 + EPS_);
  __shared__ float s1[128], s2[128];
  s1[c] = nc; s2[c] = nc * nc;
  __syncthreads();
  for (int st = 64; st > 0; st >>= 1) {
    if (c < st) { s1[c] += s1[c + st]; s2[c] += s2[c + st]; }
    __syncthreads();
  }
  float mu = s1[0] * (1.f / C_);
  float var = s2[0] * (1.f / C_) - mu * mu;
  float nh = (nc - mu) * rsqrtf(var + LNEPS_);
  float ratio = (g[c] * nh + bb[c]) / nc;
  int b = bn / N_, n = bn % N_;
  size_t r0 = ((size_t)(b * 3) * N_ + n) * C_ + c;
  nx[r0]                       = (_Float16)(x0 * ratio);
  nx[r0 + (size_t)N_ * C_]     = (_Float16)(x1 * ratio);
  nx[r0 + 2 * (size_t)N_ * C_] = (_Float16)(x2 * ratio);
}

// ---------------------------------------------------------------- LayerNorm 2
__global__ void k_ln2(const float* __restrict__ v2, const float* __restrict__ g,
                      const float* __restrict__ bb, _Float16* __restrict__ nx2) {
  int bn = blockIdx.x, c = threadIdx.x;
  int b = bn / N_, n = bn % N_;
  size_t r0 = ((size_t)(b * 3) * N_ + n) * C_ + c;
  float x0 = v2[r0], x1 = v2[r0 + (size_t)N_ * C_], x2 = v2[r0 + 2 * (size_t)N_ * C_];
  float nc = sqrtf(x0*x0 + x1*x1 + x2*x2 + EPS_);
  __shared__ float s1[128], s2[128];
  s1[c] = nc; s2[c] = nc * nc;
  __syncthreads();
  for (int st = 64; st > 0; st >>= 1) {
    if (c < st) { s1[c] += s1[c + st]; s2[c] += s2[c + st]; }
    __syncthreads();
  }
  float mu = s1[0] * (1.f / C_);
  float var = s2[0] * (1.f / C_) - mu * mu;
  float nh = (nc - mu) * rsqrtf(var + LNEPS_);
  float ratio = (g[c] * nh + bb[c]) / nc;
  nx2[r0]                       = (_Float16)(x0 * ratio);
  nx2[r0 + (size_t)N_ * C_]     = (_Float16)(x1 * ratio);
  nx2[r0 + 2 * (size_t)N_ * C_] = (_Float16)(x2 * ratio);
}

// ---------------------------------------------------------------- generic GEMM
// mode 0: scatter to QKV layout (B,H,N,192); mode 1: f16 store; mode 2: W2+residual f32.
__global__ void __launch_bounds__(256)
k_gemm(const _Float16* __restrict__ A, int lda,
       const _Float16* __restrict__ W, int cin, int cout,
       int ntiles, int mode,
       _Float16* __restrict__ outh, int ldo, int colofs,
       float* __restrict__ outf, const float* __restrict__ xres) {
  int lane = threadIdx.x & 31;
  int t = blockIdx.x * 8 + (threadIdx.x >> 5);
  if (t >= ntiles) return;
  int nct = cout >> 4;
  int ct = t % nct, mt = t / nct;
  const _Float16* abase = A + (size_t)mt * 16 * lda;
  const _Float16* wbase = W + (size_t)(ct * 16) * cin;
  v8f acc = {};
  for (int c0 = 0; c0 < cin; c0 += 32) {
    v16h a = afrag(abase, lda, c0, lane);
    v16h b = bfrag(wbase, cin, c0, lane);
    acc = wmma_f16(a, b, acc);
  }
  int col = ct * 16 + (lane & 15);
  int rbase = mt * 16 + ((lane >> 4) << 3);
  if (mode == 0) {
    int h = col / 64, oc = col % 64;
#pragma unroll
    for (int v = 0; v < 8; v++) {
      int row = rbase + v;
      int b = row / (3 * N_); int rem = row % (3 * N_);
      int d = rem / N_;       int n = rem % N_;
      outh[(((size_t)b * H_ + h) * N_ + n) * HD_ + oc * 3 + d] = (_Float16)acc[v];
    }
  } else if (mode == 1) {
#pragma unroll
    for (int v = 0; v < 8; v++) {
      int row = rbase + v;
      outh[(size_t)row * ldo + colofs + col] = (_Float16)acc[v];
    }
  } else {
#pragma unroll
    for (int v = 0; v < 8; v++) {
      int row = rbase + v;
      int b = row / (3 * N_); int rem = row % (3 * N_);
      int d = rem / N_;       int n = rem % N_;
      float val = acc[v] + xres[((size_t)b * N_ + n) * D_ + col * 3 + d];
      outf[(size_t)row * C_ + col] = val;
    }
  }
}

// ---------------------------------------------------------------- flash attention
// per block: one (b,h), 128 queries (8 waves x 16). K tile staged via TDM
// (tensor_load_to_lds, TENSORcnt), V tile staged via global_load_async_to_lds_b128
// (ASYNCcnt) then transposed in LDS for contiguous P*V B-fragments.
__global__ void __launch_bounds__(256)
k_flash(const _Float16* __restrict__ Q, const _Float16* __restrict__ Kb,
        const _Float16* __restrict__ Vb, _Float16* __restrict__ aocl) {
  __shared__ alignas(16) _Float16 Kt[32 * HD_];
  __shared__ alignas(16) _Float16 Vrow[32 * HD_];
  __shared__ alignas(16) _Float16 Vt[HD_ * 32];
  __shared__ alignas(16) _Float16 Pst[8][16 * 32];
  int tid = threadIdx.x, lane = tid & 31, w = tid >> 5;
  int bh = blockIdx.x >> 4;
  int qt = blockIdx.x & 15;
  int b = bh / H_, h = bh % H_;
  const _Float16* Qbase = Q  + (size_t)bh * N_ * HD_;
  const _Float16* Kbase = Kb + (size_t)bh * N_ * HD_;
  const _Float16* Vbase = Vb + (size_t)bh * N_ * HD_;
  int q0 = qt * 128 + w * 16;

  // LDS byte offsets (flat LDS-aperture address: low 32 bits = LDS offset)
  unsigned ktoff   = (unsigned)(uintptr_t)(&Kt[0]);
  unsigned vrowoff = (unsigned)(uintptr_t)(&Vrow[0]);

  v16h qf[6];
#pragma unroll
  for (int ch = 0; ch < 6; ch++)
    qf[ch] = afrag(Qbase + (size_t)q0 * HD_, HD_, ch * 32, lane);

  v8f o[12] = {};
  float mrow[8], lrow[8];
#pragma unroll
  for (int v = 0; v < 8; v++) { mrow[v] = -1e30f; lrow[v] = 0.f; }

  for (int kv0 = 0; kv0 < N_; kv0 += 32) {
    __syncthreads();   // previous iteration done reading Kt/Vt/Vrow

    // ---- K tile (32 x 192 f16) via Tensor Data Mover, issued by wave 0 ----
    if (w == 0) {
      unsigned long long ga =
          (unsigned long long)(uintptr_t)(Kbase + (size_t)kv0 * HD_);
      ui4 g0;
      g0.x = 1u;                                   // count=1, user descriptor
      g0.y = ktoff;                                // lds_addr (bytes)
      g0.z = (unsigned)ga;                         // global_addr[31:0]
      g0.w = (unsigned)((ga >> 32) & 0x01FFFFFFull) | (2u << 30); // addr[56:32], type=2
      ui8 g1;
      g1[0] = 0x00010000u;                         // data_size=1 (2 bytes/elem)
      g1[1] = ((unsigned)HD_ & 0xFFFFu) << 16;     // tensor_dim0[15:0]
      g1[2] = ((unsigned)HD_ >> 16) | (32u << 16); // tensor_dim0[31:16]|tensor_dim1[15:0]
      g1[3] = ((unsigned)HD_ << 16);               // tensor_dim1[31:16]|tile_dim0=192
      g1[4] = 32u;                                 // tile_dim1=32, tile_dim2=0
      g1[5] = (unsigned)HD_;                       // tensor_dim0_stride[31:0]=192
      g1[6] = 0u;
      g1[7] = 0u;
      asm volatile("tensor_load_to_lds %0, %1" :: "s"(g0), "s"(g1) : "memory");
    }

    // ---- V tile (32 x 192 f16) row-major via async global->LDS ----
#pragma unroll
    for (int j = 0; j < 3; j++) {
      int id = tid + 256 * j;                      // 0..767 chunks of 8 halfs
      int r = id / 24, c8 = (id % 24) * 8;
      unsigned lo = vrowoff + (unsigned)((r * HD_ + c8) * 2);
      unsigned long long ga =
          (unsigned long long)(uintptr_t)(Vbase + (size_t)(kv0 + r) * HD_ + c8);
      asm volatile("global_load_async_to_lds_b128 %0, %1, off"
                   :: "v"(lo), "v"(ga) : "memory");
    }
    asm volatile("s_wait_asynccnt 0x0" ::: "memory");
    if (w == 0) __builtin_amdgcn_s_wait_tensorcnt(0);
    __syncthreads();

    // ---- transpose Vrow -> Vt in LDS ----
#pragma unroll
    for (int j = 0; j < 3; j++) {
      int id = tid + 256 * j;
      int r = id / 24, c8 = (id % 24) * 8;
      h8 vv = *(const h8*)(&Vrow[r * HD_ + c8]);
#pragma unroll
      for (int ii = 0; ii < 8; ii++) Vt[(c8 + ii) * 32 + r] = vv[ii];
    }
    __syncthreads();

    // ---- S = Q K^T (two 16x16 tiles over the 32 keys) ----
    v8f s0 = {}, s1 = {};
#pragma unroll
    for (int ch = 0; ch < 6; ch++) {
      v16h bk0 = bfrag(Kt, HD_, ch * 32, lane);
      v16h bk1 = bfrag(Kt + 16 * HD_, HD_, ch * 32, lane);
      s0 = wmma_f16(qf[ch], bk0, s0);
      s1 = wmma_f16(qf[ch], bk1, s1);
    }

    float p0[8], p1[8], alpha[8];
#pragma unroll
    for (int v = 0; v < 8; v++) {
      float a = s0[v] * 0.125f, c = s1[v] * 0.125f;
      float t = fmaxf(a, c);
      t = fmaxf(t, __shfl_xor(t, 1, 32));
      t = fmaxf(t, __shfl_xor(t, 2, 32));
      t = fmaxf(t, __shfl_xor(t, 4, 32));
      t = fmaxf(t, __shfl_xor(t, 8, 32));
      float mnew = fmaxf(mrow[v], t);
      float al = __expf(mrow[v] - mnew);
      float e0 = __expf(a - mnew), e1 = __expf(c - mnew);
      float rs = e0 + e1;
      rs += __shfl_xor(rs, 1, 32);
      rs += __shfl_xor(rs, 2, 32);
      rs += __shfl_xor(rs, 4, 32);
      rs += __shfl_xor(rs, 8, 32);
      lrow[v] = lrow[v] * al + rs;
      mrow[v] = mnew;
      alpha[v] = al; p0[v] = e0; p1[v] = e1;
    }
#pragma unroll
    for (int i = 0; i < 12; i++)
#pragma unroll
      for (int v = 0; v < 8; v++) o[i][v] *= alpha[v];

    {
      int half = lane >> 4, cc = lane & 15;
#pragma unroll
      for (int v = 0; v < 8; v++) {
        int row = half * 8 + v;
        Pst[w][row * 32 + cc]      = (_Float16)p0[v];
        Pst[w][row * 32 + 16 + cc] = (_Float16)p1[v];
      }
    }
    __syncthreads();
    v16h pa = afrag(&Pst[w][0], 32, 0, lane);
#pragma unroll
    for (int dc = 0; dc < 12; dc++) {
      v16h bv = bfrag(Vt + dc * 16 * 32, 32, 0, lane);
      o[dc] = wmma_f16(pa, bv, o[dc]);
    }
  }

  float rinv[8];
#pragma unroll
  for (int v = 0; v < 8; v++) rinv[v] = 1.f / lrow[v];
#pragma unroll
  for (int dc = 0; dc < 12; dc++) {
    int j = dc * 16 + (lane & 15);
    int oc = j / 3, d = j % 3, f = h * 64 + oc;
#pragma unroll
    for (int v = 0; v < 8; v++) {
      int n = q0 + ((lane >> 4) << 3) + v;
      aocl[(((size_t)b * 3 + d) * N_ + n) * D_ + f] = (_Float16)(o[dc][v] * rinv[v]);
    }
  }
}

// ---------------------------------------------------------------- knn gather + conv1 + leaky + k-meanpool
__global__ void __launch_bounds__(256)
k_knn(const _Float16* __restrict__ nx, const int* __restrict__ knn,
      const _Float16* __restrict__ W1, const _Float16* __restrict__ U1,
      _Float16* __restrict__ cat) {
  int lane = threadIdx.x & 31;
  int t = blockIdx.x * 8 + (threadIdx.x >> 5);
  if (t >= B_ * (N_ / 2) * 8) return;
  int ct = t & 7; int rem = t >> 3;
  int np = rem % (N_ / 2); int b = rem / (N_ / 2);
  int n0 = np * 2;
  int row = lane & 15;
  int nl = row >> 3, k = row & 7;
  int n = n0 + nl;
  int idx = knn[(size_t)b * KNN_ * N_ + (size_t)k * N_ + n];
  int bp = idx / N_, npn = idx % N_;
  const _Float16* wb = W1 + (size_t)(ct * 16) * 256;
  const _Float16* ub = U1 + (size_t)(ct * 16) * 256;
  int half8 = (lane >> 4) << 3;
  v8f p[3] = {}, u[3] = {};
  __builtin_prefetch(nx + ((size_t)(bp * 3) * N_ + npn) * C_, 0, 0);
  for (int chunk = 0; chunk < 8; chunk++) {
    v16h bw = bfrag(wb, 256, chunk * 32, lane);
    v16h bu = bfrag(ub, 256, chunk * 32, lane);
#pragma unroll
    for (int d = 0; d < 3; d++) {
      const _Float16* nb  = nx + ((size_t)(bp * 3 + d) * N_ + npn) * C_;
      const _Float16* ctr = nx + ((size_t)(b  * 3 + d) * N_ + n)   * C_;
      v16h a;
      if (chunk < 4) {
        int c0 = chunk * 32 + half8;
        h8 a0 = *(const h8*)(nb + c0)      - *(const h8*)(ctr + c0);
        h8 a1 = *(const h8*)(nb + c0 + 16) - *(const h8*)(ctr + c0 + 16);
        a = cat16(a0, a1);
      } else {
        int c0 = (chunk - 4) * 32 + half8;
        a = cat16(*(const h8*)(ctr + c0), *(const h8*)(ctr + c0 + 16));
      }
      p[d] = wmma_f16(a, bw, p[d]);
      u[d] = wmma_f16(a, bu, u[d]);
    }
  }
  int col = ct * 16 + (lane & 15);
  float acc[3] = {0.f, 0.f, 0.f};
#pragma unroll
  for (int v = 0; v < 8; v++) {
    float dot = p[0][v]*u[0][v] + p[1][v]*u[1][v] + p[2][v]*u[2][v];
    float dsq = u[0][v]*u[0][v] + u[1][v]*u[1][v] + u[2][v]*u[2][v];
    float coef = dot / (dsq + EPS_);
    bool pos = dot >= 0.f;
#pragma unroll
    for (int d = 0; d < 3; d++) {
      float pd = p[d][v];
      float od = NS_ * pd + (1.f - NS_) * (pos ? pd : (pd - coef * u[d][v]));
      acc[d] += od;
    }
  }
  int nOut = n0 + (lane >> 4);
#pragma unroll
  for (int d = 0; d < 3; d++)
    cat[((size_t)(b * 3 + d) * N_ + nOut) * 256 + 128 + col] = (_Float16)(acc[d] * 0.125f);
}

// ---------------------------------------------------------------- generic VN-LeakyReLU conv
__global__ void __launch_bounds__(256)
k_leaky(const _Float16* __restrict__ in, int cin,
        const _Float16* __restrict__ W, const _Float16* __restrict__ U,
        int cout, int ntiles, int mode,
        _Float16* __restrict__ outh, int ldo, int colofs,
        const float* __restrict__ vnx2, float* __restrict__ outf) {
  int lane = threadIdx.x & 31;
  int t = blockIdx.x * 8 + (threadIdx.x >> 5);
  if (t >= ntiles) return;
  int nct = cout >> 4;
  int ct = t % nct; int rem = t / nct;
  int nt = rem % (N_ / 16); int b = rem / (N_ / 16);
  int n0 = nt * 16;
  const _Float16* wb = W + (size_t)(ct * 16) * cin;
  const _Float16* ub = U + (size_t)(ct * 16) * cin;
  v8f p[3] = {}, u[3] = {};
  for (int c0 = 0; c0 < cin; c0 += 32) {
    v16h bw = bfrag(wb, cin, c0, lane);
    v16h bu = bfrag(ub, cin, c0, lane);
#pragma unroll
    for (int d = 0; d < 3; d++) {
      const _Float16* ab = in + ((size_t)(b * 3 + d) * N_ + n0) * cin;
      v16h a = afrag(ab, cin, c0, lane);
      p[d] = wmma_f16(a, bw, p[d]);
      u[d] = wmma_f16(a, bu, u[d]);
    }
  }
  int col = ct * 16 + (lane & 15);
#pragma unroll
  for (int v = 0; v < 8; v++) {
    int n = n0 + ((lane >> 4) << 3) + v;
    float dot = p[0][v]*u[0][v] + p[1][v]*u[1][v] + p[2][v]*u[2][v];
    float dsq = u[0][v]*u[0][v] + u[1][v]*u[1][v] + u[2][v]*u[2][v];
    float coef = dot / (dsq + EPS_);
    bool pos = dot >= 0.f;
#pragma unroll
    for (int d = 0; d < 3; d++) {
      float pd = p[d][v];
      float od = NS_ * pd + (1.f - NS_) * (pos ? pd : (pd - coef * u[d][v]));
      if (mode == 0) {
        outh[((size_t)(b * 3 + d) * N_ + n) * ldo + colofs + col] = (_Float16)od;
      } else {
        float val = vnx2[((size_t)(b * 3 + d) * N_ + n) * C_ + col] + od;
        outf[((size_t)b * N_ + n) * D_ + col * 3 + d] = val;
      }
    }
  }
}

// ---------------------------------------------------------------- host
extern "C" void kernel_launch(void* const* d_in, const int* in_sizes, int n_in,
                              void* d_out, int out_size, void* d_ws, size_t ws_size,
                              hipStream_t stream) {
  (void)in_sizes; (void)n_in; (void)out_size; (void)ws_size;
  const float* x  = (const float*)d_in[0];
  const int* knn  = (const int*)d_in[1];
  const float* g1 = (const float*)d_in[2];
  const float* b1 = (const float*)d_in[3];
  const float* g2 = (const float*)d_in[4];
  const float* b2 = (const float*)d_in[5];
  const float* Wq = (const float*)d_in[6];
  const float* Wk = (const float*)d_in[7];
  const float* Wv = (const float*)d_in[8];
  const float* Wo = (const float*)d_in[9];
  const float* W1 = (const float*)d_in[10];
  const float* U1 = (const float*)d_in[11];
  const float* W2 = (const float*)d_in[12];
  const float* W3 = (const float*)d_in[13];
  const float* U3 = (const float*)d_in[14];
  const float* W4 = (const float*)d_in[15];
  const float* U4 = (const float*)d_in[16];
  float* out = (float*)d_out;

  char* ws = (char*)d_ws;
  size_t off = 0;
  auto alloc = [&](size_t bytes) -> void* {
    off = (off + 255) & ~(size_t)255;
    void* p = ws + off; off += bytes; return p;
  };
  const size_t PC = (size_t)B_ * 3 * N_ * C_;        // 3,145,728
  const size_t QKV = (size_t)B_ * H_ * N_ * HD_;     // 9,437,184

  _Float16* wqh = (_Float16*)alloc(49152 * 2);
  _Float16* wkh = (_Float16*)alloc(49152 * 2);
  _Float16* wvh = (_Float16*)alloc(49152 * 2);
  _Float16* woh = (_Float16*)alloc(49152 * 2);
  _Float16* w1h = (_Float16*)alloc(32768 * 2);
  _Float16* u1h = (_Float16*)alloc(32768 * 2);
  _Float16* w2h = (_Float16*)alloc(32768 * 2);
  _Float16* w3h = (_Float16*)alloc(32768 * 2);
  _Float16* u3h = (_Float16*)alloc(32768 * 2);
  _Float16* w4h = (_Float16*)alloc(32768 * 2);
  _Float16* u4h = (_Float16*)alloc(32768 * 2);
  _Float16* nx   = (_Float16*)alloc(PC * 2);
  _Float16* Qb   = (_Float16*)alloc(QKV * 2);
  _Float16* Kb   = (_Float16*)alloc(QKV * 2);
  _Float16* Vb   = (_Float16*)alloc(QKV * 2);
  _Float16* aocl = (_Float16*)alloc((size_t)B_ * 3 * N_ * D_ * 2);
  _Float16* cat  = (_Float16*)alloc((size_t)B_ * 3 * N_ * 256 * 2);
  float*    vnx2 = (float*)   alloc(PC * 4);
  _Float16* nx2  = (_Float16*)alloc(PC * 2);
  _Float16* h3   = (_Float16*)alloc((size_t)B_ * 3 * N_ * 256 * 2);

  k_cvt<<<96, 256, 0, stream>>>(Wq, wqh, 49152);
  k_cvt<<<96, 256, 0, stream>>>(Wk, wkh, 49152);
  k_cvt<<<96, 256, 0, stream>>>(Wv, wvh, 49152);
  k_cvt<<<96, 256, 0, stream>>>(Wo, woh, 49152);
  k_cvt<<<64, 256, 0, stream>>>(W1, w1h, 32768);
  k_cvt<<<64, 256, 0, stream>>>(U1, u1h, 32768);
  k_cvt<<<64, 256, 0, stream>>>(W2, w2h, 32768);
  k_cvt<<<64, 256, 0, stream>>>(W3, w3h, 32768);
  k_cvt<<<64, 256, 0, stream>>>(U3, u3h, 32768);
  k_cvt<<<64, 256, 0, stream>>>(W4, w4h, 32768);
  k_cvt<<<64, 256, 0, stream>>>(U4, u4h, 32768);

  // LayerNorm 1
  k_ln1<<<B_ * N_, 128, 0, stream>>>(x, g1, b1, nx);

  // Q/K/V projections: rows = 24576, tiles = 1536*24 = 36864
  k_gemm<<<4608, 256, 0, stream>>>(nx, C_, wqh, C_, D_, 36864, 0, Qb, 0, 0, nullptr, nullptr);
  k_gemm<<<4608, 256, 0, stream>>>(nx, C_, wkh, C_, D_, 36864, 0, Kb, 0, 0, nullptr, nullptr);
  k_gemm<<<4608, 256, 0, stream>>>(nx, C_, wvh, C_, D_, 36864, 0, Vb, 0, 0, nullptr, nullptr);

  // flash attention: 24 (b,h) * 16 q-blocks
  k_flash<<<384, 256, 0, stream>>>(Qb, Kb, Vb, aocl);

  // Wo: (128 x 384) -> cat cols 0..127 ; tiles = 1536*8 = 12288
  k_gemm<<<1536, 256, 0, stream>>>(aocl, D_, woh, D_, C_, 12288, 1, cat, 256, 0, nullptr, nullptr);

  // knn conv1 -> cat cols 128..255 ; waves = 4*1024*8 = 32768
  k_knn<<<4096, 256, 0, stream>>>(nx, knn, w1h, u1h, cat);

  // W2 + residual -> vnx2 (f32 channel-last) ; tiles = 12288
  k_gemm<<<1536, 256, 0, stream>>>(cat, 256, w2h, 256, C_, 12288, 2, nullptr, 0, 0, vnx2, x);

  // LayerNorm 2
  k_ln2<<<B_ * N_, 128, 0, stream>>>(vnx2, g2, b2, nx2);

  // conv3: 128 -> 256, leaky ; tiles = 4*128*16 = 8192
  k_leaky<<<1024, 256, 0, stream>>>(nx2, C_, w3h, u3h, 256, 8192, 0, h3, 256, 0, nullptr, nullptr);

  // conv4: 256 -> 128, leaky + residual + final store ; tiles = 4*128*8 = 4096
  k_leaky<<<512, 256, 0, stream>>>(h3, 256, w4h, u4h, C_, 4096, 1, nullptr, 0, 0, vnx2, out);
}
